// CausalSelfAttention_56573309223593
// MI455X (gfx1250) — matrix operand order repaired and checked by reading, not compiled, not run on hardware
//
#include <hip/hip_runtime.h>
#include <stdint.h>

#define BB 2
#define SS 2048
#define DD 2048
#define HH 16
#define DHD 128

typedef __attribute__((ext_vector_type(16))) __bf16 v16bf;
typedef __attribute__((ext_vector_type(8)))  float  v8f;

union Frag {
    v16bf v;
    uint4 q[2];
};

__device__ inline v8f zero8() {
    v8f z;
#pragma unroll
    for (int i = 0; i < 8; ++i) z[i] = 0.0f;
    return z;
}

__device__ inline uint16_t f2bf(float f) {
    uint32_t u = __float_as_uint(f);
    u += 0x7FFFu + ((u >> 16) & 1u);   // round to nearest even
    return (uint16_t)(u >> 16);
}

// ---- CDNA5 async copy: global -> LDS, tracked by ASYNCcnt ----------------
__device__ inline void async_ld128(uint32_t ldsaddr, const uint16_t* gaddr) {
    asm volatile("global_load_async_to_lds_b128 %0, %1, off"
                 :: "v"(ldsaddr), "v"(gaddr)
                 : "memory");
}
__device__ inline void wait_async0() {
    asm volatile("s_wait_asynccnt 0x0" ::: "memory");
}
__device__ inline uint32_t lds_off(const void* p) {
    return (uint32_t)(uintptr_t)p;   // low 32 bits of flat addr = LDS offset
}

// ---------------------------------------------------------------- converts
__global__ void k_f32_to_bf16(const float* __restrict__ in,
                              uint16_t* __restrict__ out, int n) {
    int i = blockIdx.x * blockDim.x + threadIdx.x;
    if (i < n) out[i] = f2bf(in[i]);
}

// in is K x N row-major; out is N x K (bf16): out[n*K + k] = in[k*N + n]
__global__ void k_transpose_bf16(const float* __restrict__ in,
                                 uint16_t* __restrict__ out, int K, int N) {
    int i = blockIdx.x * blockDim.x + threadIdx.x;
    if (i >= K * N) return;
    int n = i / K;
    int k = i - n * K;
    out[i] = f2bf(in[(long)k * N + n]);
}

// A-fragment (16x32 bf16) direct from global (used once for Q in attention)
__device__ inline Frag load_a_g(const uint16_t* __restrict__ A, long lda,
                                int row, int k0, int lane) {
    Frag f;
    const uint16_t* p = A + (long)row * lda + k0 + 8 * (lane >> 4);
    f.q[0] = *(const uint4*)(p);
    f.q[1] = *(const uint4*)(p + 16);
    return f;
}

// ================================================================ GEMM core
// C[128x128] per block, 8 waves x (16 rows x 128 cols), bf16 WMMA, fp32 acc.
// A: [M][K] row-major bf16, Bt: [N][K] bf16. Double-buffered async LDS tiles.
#define TS 40   // padded LDS row stride (bf16 elements) for 32-wide K slices

template <typename EPI>
__device__ inline void gemm_core(const uint16_t* __restrict__ A,
                                 const uint16_t* __restrict__ Bt,
                                 int rowBase, int colBase, EPI epilogue) {
    __shared__ __align__(16) uint16_t As[2][128 * TS];
    __shared__ __align__(16) uint16_t Bs[2][128 * TS];

    const int tid  = threadIdx.x;
    const int lane = tid & 31;
    const int wave = tid >> 5;
    const int grp  = lane >> 4;

    v8f acc[8];
#pragma unroll
    for (int j = 0; j < 8; ++j) acc[j] = zero8();

    // cooperative stage of one 128x32 A tile + 128x32 B tile into buffer b
    auto stage = [&](int b, int k0) {
#pragma unroll
        for (int i = 0; i < 2; ++i) {
            int id  = tid + 256 * i;
            int row = id >> 2, c = id & 3;           // 4 x 16B chunks per row
            async_ld128(lds_off(&As[b][row * TS + 8 * c]),
                        A + (long)(rowBase + row) * DD + k0 + 8 * c);
            async_ld128(lds_off(&Bs[b][row * TS + 8 * c]),
                        Bt + (long)(colBase + row) * DD + k0 + 8 * c);
        }
    };

    int buf = 0;
    stage(0, 0);
    wait_async0();
    __syncthreads();

    for (int k0 = 0; k0 < DD; k0 += 32) {
        if (k0 + 32 < DD) stage(buf ^ 1, k0 + 32);

        // batch all fragment loads, then issue 8 back-to-back WMMAs
        Frag a;
        const uint16_t* ap = &As[buf][(wave * 16 + (lane & 15)) * TS + 8 * grp];
        a.q[0] = *(const uint4*)(ap);
        a.q[1] = *(const uint4*)(ap + 16);
        Frag bfr[8];
#pragma unroll
        for (int j = 0; j < 8; ++j) {
            const uint16_t* bp = &Bs[buf][(16 * j + (lane & 15)) * TS + 8 * grp];
            bfr[j].q[0] = *(const uint4*)(bp);
            bfr[j].q[1] = *(const uint4*)(bp + 16);
        }
#pragma unroll
        for (int j = 0; j < 8; ++j)
            acc[j] = __builtin_amdgcn_wmma_f32_16x16x32_bf16(
                false, a.v, false, bfr[j].v, (short)0, acc[j], false, false);

        wait_async0();      // my async writes for buf^1 landed
        __syncthreads();    // everyone's reads of buf + writes of buf^1 done
        buf ^= 1;
    }
    epilogue(acc, lane, wave, grp);
}

// ---------------------------------------------------------------- QKV GEMM
__global__ void __launch_bounds__(256) k_gemm_qkv(
    const uint16_t* __restrict__ Xb,     // [B*S][D]
    const uint16_t* __restrict__ WqkvT,  // [3D][D]
    const float*    __restrict__ bqkv,
    uint16_t* __restrict__ qb,           // [B*H][S][DH]
    uint16_t* __restrict__ kb,           // [B*H][S][DH]
    uint16_t* __restrict__ vTb)          // [B*H][DH][S]
{
    const int rowBase = blockIdx.y * 128;
    const int colBase = blockIdx.x * 128;
    gemm_core(Xb, WqkvT, rowBase, colBase,
        [&](v8f* acc, int lane, int wave, int grp) {
#pragma unroll
            for (int j = 0; j < 8; ++j) {
                int c  = colBase + 16 * j + (lane & 15);
                int wh = c / DD;          // 0=Q 1=K 2=V
                int cc = c - wh * DD;
                int h  = cc / DHD;
                int dh = cc - h * DHD;
                float bias = bqkv[c];
#pragma unroll
                for (int r = 0; r < 8; ++r) {
                    int m  = rowBase + wave * 16 + r + 8 * grp;
                    int b_ = m / SS;
                    int s  = m - b_ * SS;
                    long bh = (long)b_ * HH + h;
                    uint16_t val = f2bf(acc[j][r] + bias);
                    if (wh == 0)      qb[(bh * SS + s) * DHD + dh] = val;
                    else if (wh == 1) kb[(bh * SS + s) * DHD + dh] = val;
                    else              vTb[(bh * DHD + dh) * SS + s] = val;
                }
            }
        });
}

// ---------------------------------------------------------------- proj GEMM
__global__ void __launch_bounds__(256) k_gemm_proj(
    const uint16_t* __restrict__ Ab,      // [B*S][D]
    const uint16_t* __restrict__ WprojT,  // [D][D]
    const float*    __restrict__ bproj,
    float* __restrict__ out)              // [B*S][D] fp32
{
    const int rowBase = blockIdx.y * 128;
    const int colBase = blockIdx.x * 128;
    gemm_core(Ab, WprojT, rowBase, colBase,
        [&](v8f* acc, int lane, int wave, int grp) {
#pragma unroll
            for (int j = 0; j < 8; ++j) {
                int c = colBase + 16 * j + (lane & 15);
                float bias = bproj[c];
#pragma unroll
                for (int r = 0; r < 8; ++r) {
                    int m = rowBase + wave * 16 + r + 8 * grp;
                    out[(long)m * DD + c] = acc[j][r] + bias;
                }
            }
        });
}

// ---------------------------------------------------------------- attention
#define KST 136   // padded LDS stride for K tile rows (128 dh + pad)
#define VST 40    // padded LDS stride for V tile rows (32 keys + pad)

__global__ void __launch_bounds__(256) k_attn(
    const uint16_t* __restrict__ qb,     // [B*H][S][DH]
    const uint16_t* __restrict__ kb,     // [B*H][S][DH]
    const uint16_t* __restrict__ vTb,    // [B*H][DH][S]
    uint16_t* __restrict__ attnb)        // [B*S][D]
{
    __shared__ __align__(16) uint16_t Ks[2][32 * KST];
    __shared__ __align__(16) uint16_t Vs[2][128 * VST];
    __shared__ __align__(16) uint16_t pshared[8][16 * 40];

    const int tid  = threadIdx.x;
    const int lane = tid & 31;
    const int wave = tid >> 5;
    const int grp  = lane >> 4;
    const int bh   = blockIdx.y;
    const int qBase = blockIdx.x * 128 + wave * 16;

    const uint16_t* qh = qb  + (long)bh * SS * DHD;
    const uint16_t* kh = kb  + (long)bh * SS * DHD;
    const uint16_t* vh = vTb + (long)bh * DHD * SS;
    const float scale = 0.08838834764831845f;   // 1/sqrt(128)

    Frag qf[4];
#pragma unroll
    for (int c = 0; c < 4; ++c)
        qf[c] = load_a_g(qh, DHD, qBase + (lane & 15), 32 * c, lane);

    v8f acc[8];
#pragma unroll
    for (int j = 0; j < 8; ++j) acc[j] = zero8();
    float m_i[8], l_i[8];
#pragma unroll
    for (int r = 0; r < 8; ++r) { m_i[r] = -1e30f; l_i[r] = 0.0f; }

    uint16_t* pw = &pshared[wave][0];

    // cooperative stage of 32-key K tile (32x128) and V tile (128x32)
    auto stage = [&](int b, int kc) {
#pragma unroll
        for (int i = 0; i < 2; ++i) {
            int id = tid + 256 * i;
            int kr = id >> 4, kck = id & 15;         // K: 16 x 16B chunks/row
            async_ld128(lds_off(&Ks[b][kr * KST + 8 * kck]),
                        kh + (long)(kc + kr) * DHD + 8 * kck);
            int vr = id >> 2, vc = id & 3;           // V: 4 x 16B chunks/row
            async_ld128(lds_off(&Vs[b][vr * VST + 8 * vc]),
                        vh + (long)vr * SS + kc + 8 * vc);
        }
    };

    const int nch = blockIdx.x * 4 + 4;   // chunks covering keys 0..blockMax
    int buf = 0;
    stage(0, 0);
    wait_async0();
    __syncthreads();

    for (int ic = 0; ic < nch; ++ic) {
        const int kc = ic * 32;
        if (ic + 1 < nch) stage(buf ^ 1, kc + 32);

        if (kc <= qBase + 15) {        // causal: this wave still needs chunk
            // ---- scores S = Q K^T : load all 8 K-frags, then 8 WMMAs
            Frag kf[2][4];
#pragma unroll
            for (int t = 0; t < 2; ++t)
#pragma unroll
                for (int c = 0; c < 4; ++c) {
                    const uint16_t* bp =
                        &Ks[buf][(16 * t + (lane & 15)) * KST + 32 * c + 8 * grp];
                    kf[t][c].q[0] = *(const uint4*)(bp);
                    kf[t][c].q[1] = *(const uint4*)(bp + 16);
                }
            v8f st[2];
#pragma unroll
            for (int t = 0; t < 2; ++t) st[t] = zero8();
#pragma unroll
            for (int t = 0; t < 2; ++t)
#pragma unroll
                for (int c = 0; c < 4; ++c)
                    st[t] = __builtin_amdgcn_wmma_f32_16x16x32_bf16(
                        false, qf[c].v, false, kf[t][c].v, (short)0, st[t],
                        false, false);
            // ---- scale + causal mask
#pragma unroll
            for (int t = 0; t < 2; ++t) {
                int key = kc + 16 * t + (lane & 15);
#pragma unroll
                for (int r = 0; r < 8; ++r) {
                    int qrow = qBase + r + 8 * grp;
                    float v = st[t][r] * scale;
                    st[t][r] = (key > qrow) ? -1e30f : v;
                }
            }
            // ---- online softmax (row stats live in 16-lane half-groups)
            float mnew[8], alpha[8];
#pragma unroll
            for (int r = 0; r < 8; ++r) {
                float rm = fmaxf(st[0][r], st[1][r]);
#pragma unroll
                for (int mx = 1; mx <= 8; mx <<= 1)
                    rm = fmaxf(rm, __shfl_xor(rm, mx, 32));
                mnew[r]  = fmaxf(m_i[r], rm);
                alpha[r] = __expf(m_i[r] - mnew[r]);
                m_i[r]   = mnew[r];
            }
#pragma unroll
            for (int r = 0; r < 8; ++r) {
                float p0 = __expf(st[0][r] - mnew[r]);
                float p1 = __expf(st[1][r] - mnew[r]);
                st[0][r] = p0; st[1][r] = p1;
                float s2 = p0 + p1;
#pragma unroll
                for (int mx = 1; mx <= 8; mx <<= 1)
                    s2 += __shfl_xor(s2, mx, 32);
                l_i[r] = l_i[r] * alpha[r] + s2;
            }
#pragma unroll
            for (int j = 0; j < 8; ++j)
#pragma unroll
                for (int r = 0; r < 8; ++r) acc[j][r] *= alpha[r];

            // ---- P (D-layout) -> wave-private LDS -> A-fragment layout
#pragma unroll
            for (int t = 0; t < 2; ++t)
#pragma unroll
                for (int r = 0; r < 8; ++r)
                    pw[(r + 8 * grp) * 40 + 16 * t + (lane & 15)] = f2bf(st[t][r]);
            Frag pa;
            {
                const uint16_t* pp = pw + (lane & 15) * 40 + 8 * grp;
                pa.q[0] = *(const uint4*)(pp);
                pa.q[1] = *(const uint4*)(pp + 16);
            }
            // ---- O += P V : load all 8 V-frags, then 8 WMMAs
            Frag vf[8];
#pragma unroll
            for (int j = 0; j < 8; ++j) {
                const uint16_t* vp =
                    &Vs[buf][(16 * j + (lane & 15)) * VST + 8 * grp];
                vf[j].q[0] = *(const uint4*)(vp);
                vf[j].q[1] = *(const uint4*)(vp + 16);
            }
#pragma unroll
            for (int j = 0; j < 8; ++j)
                acc[j] = __builtin_amdgcn_wmma_f32_16x16x32_bf16(
                    false, pa.v, false, vf[j].v, (short)0, acc[j], false, false);
        }
        wait_async0();
        __syncthreads();
        buf ^= 1;
    }

    // ---- normalize + store bf16 [B*S][D]
    int b_ = bh / HH, h = bh - b_ * HH;
#pragma unroll
    for (int r = 0; r < 8; ++r) {
        float inv = 1.0f / l_i[r];
        long row = (long)b_ * SS + (qBase + r + 8 * grp);
#pragma unroll
        for (int j = 0; j < 8; ++j) {
            int col = h * DHD + 16 * j + (lane & 15);
            attnb[row * DD + col] = f2bf(acc[j][r] * inv);
        }
    }
}

// ---------------------------------------------------------------- launch
extern "C" void kernel_launch(void* const* d_in, const int* in_sizes, int n_in,
                              void* d_out, int out_size, void* d_ws, size_t ws_size,
                              hipStream_t stream) {
    (void)in_sizes; (void)n_in; (void)out_size; (void)ws_size;
    const float* x     = (const float*)d_in[0];
    const float* Wqkv  = (const float*)d_in[1];
    const float* bqkv  = (const float*)d_in[2];
    const float* Wproj = (const float*)d_in[3];
    const float* bproj = (const float*)d_in[4];
    float* out = (float*)d_out;

    char* ws = (char*)d_ws;
    size_t off = 0;
    auto take = [&](size_t bytes) -> char* {
        char* p = ws + off;
        off += (bytes + 255) & ~(size_t)255;
        return p;
    };
    uint16_t* xb     = (uint16_t*)take((size_t)BB * SS * DD * 2);
    uint16_t* wqkvT  = (uint16_t*)take((size_t)3 * DD * DD * 2);
    uint16_t* wprojT = (uint16_t*)take((size_t)DD * DD * 2);
    uint16_t* qbuf   = (uint16_t*)take((size_t)BB * HH * SS * DHD * 2);
    uint16_t* kbuf   = (uint16_t*)take((size_t)BB * HH * SS * DHD * 2);
    uint16_t* vTbuf  = (uint16_t*)take((size_t)BB * HH * DHD * SS * 2);
    uint16_t* attnb  = (uint16_t*)take((size_t)BB * SS * DD * 2);

    const int nX = BB * SS * DD;
    hipLaunchKernelGGL(k_f32_to_bf16, dim3((nX + 255) / 256), dim3(256), 0, stream,
                       x, xb, nX);
    const int nWq = DD * 3 * DD;
    hipLaunchKernelGGL(k_transpose_bf16, dim3((nWq + 255) / 256), dim3(256), 0, stream,
                       Wqkv, wqkvT, DD, 3 * DD);
    const int nWp = DD * DD;
    hipLaunchKernelGGL(k_transpose_bf16, dim3((nWp + 255) / 256), dim3(256), 0, stream,
                       Wproj, wprojT, DD, DD);

    hipLaunchKernelGGL(k_gemm_qkv, dim3(3 * DD / 128, BB * SS / 128), dim3(256), 0, stream,
                       xb, wqkvT, bqkv, qbuf, kbuf, vTbuf);
    hipLaunchKernelGGL(k_attn, dim3(SS / 128, BB * HH), dim3(256), 0, stream,
                       qbuf, kbuf, vTbuf, attnb);
    hipLaunchKernelGGL(k_gemm_proj, dim3(DD / 128, BB * SS / 128), dim3(256), 0, stream,
                       attnb, wprojT, bproj, out);
}
